// CAutomaton_25409026523269
// MI455X (gfx1250) — compile-verified, matrix operand order checked by reading, not run on hardware
//
#include <hip/hip_runtime.h>

// ---------------------------------------------------------------------------
// NCA step on MI455X (gfx1250).
//   - x halo tile staged to LDS once per 64-pixel strip via TDM
//     (tensor_load_to_lds, LDS padding gives channel stride 66) + 72 scalar
//     loads for the circular-wrap edge columns.
//   - All three matmuls (3x3 conv as im2col GEMM, 48->96, 96->12) on
//     v_wmma_f32_16x16x32_bf16 with f32 accumulation.
//   - Roofline: ~302 MB HBM traffic (~13 us @ 23.3 TB/s) vs ~46 GFLOP of
//     bf16 WMMA -> roughly balanced; fp32 WMMA (16x16x4) would be ~8x slower.
// ---------------------------------------------------------------------------

typedef __attribute__((ext_vector_type(16))) __bf16        v16bf;
typedef __attribute__((ext_vector_type(8)))  float         v8f;
typedef __attribute__((ext_vector_type(4)))  unsigned int  u32x4;
typedef __attribute__((ext_vector_type(8)))  int           i32x8;
typedef __attribute__((ext_vector_type(4)))  int           i32x4;

#define NS    12      // states (input channels)
#define PC    48      // perception channels
#define HID   96      // hidden features
#define KP    128     // perception K padded (108 -> 128)
#define K1    64      // stage-1 K padded (48 -> 64)
#define K2    96      // stage-2 K (exact, 3x32)
#define HW    512
#define WAVES 4       // waves per block (128 threads)
#define STRIP 64      // pixels per block per group (4 waves x 16)
#define XCOLS 66      // strip + 2 halo columns
#define GRP   4       // strips per block

__device__ __forceinline__ unsigned short bfbits(float f) {
  union { __bf16 h; unsigned short s; } u;
  u.h = (__bf16)f;
  return u.s;
}

// 16x32 bf16 WMMA fragment from an LDS matrix stored row-major [rows][stride]
// (stride even). Used for A ([M][K]) and B stored transposed as [N][K].
// Layout per CDNA5 ISA 7.12.2 (16-bit A-matrix).
__device__ __forceinline__ v16bf load_frag(const unsigned short* mat, int row,
                                           int stride, int kc, int lane) {
  union { v16bf v; unsigned int u[8]; } f;
  const unsigned int* p = (const unsigned int*)(mat + row * stride);
  const int kb = kc * 32 + ((lane >> 4) << 3);
  #pragma unroll
  for (int i = 0; i < 8; ++i) {
    const int k = (i < 4) ? (kb + i * 2) : (kb + 16 + (i - 4) * 2);
    f.u[i] = p[k >> 1];
  }
  return f.v;
}

__device__ __forceinline__ v8f wmma_bf16(v16bf a, v16bf b, v8f c) {
  return __builtin_amdgcn_wmma_f32_16x16x32_bf16(
      false, a, false, b, (short)0, c, false, false);
}

// TDM: load one halo row (64 cols x 12 channels, f32) into LDS with a
// 2-dword pad every 64 dwords so channels land at stride 66 floats.
// D# per CDNA5 ISA 08_async_tensor 8.3/8.4 (2D tile, groups 2/3 zero).
__device__ __forceinline__ void tdm_load_row(const float* gaddr, unsigned lds_off) {
  const unsigned long long ga = (unsigned long long)(const void*)gaddr;
  u32x4 g0;
  g0[0] = 1u;                                        // count=1, user mode
  g0[1] = lds_off;                                   // lds_addr (bytes)
  g0[2] = (unsigned)(ga & 0xFFFFFFFFull);            // global_addr[31:0]
  g0[3] = (unsigned)((ga >> 32) & 0x01FFFFFFull)     // global_addr[56:32]
        | 0x80000000u;                               // type=2 -> bits[127:126]=10b
  i32x8 g1;
  g1[0] = (2 << 16)      // data_size = 4B
        | (1 << 20)      // pad_enable
        | (5 << 22)      // pad_interval: 64 dwords
        | (1 << 25);     // pad_amount: 2 dwords
  g1[1] = (HW << 16);    // atomic_addr=0 | tensor_dim0[15:0] = 512
  g1[2] = (NS << 16);    // tensor_dim0 hi=0 | tensor_dim1[15:0] = 12
  g1[3] = (STRIP << 16); // tensor_dim1 hi=0 | tile_dim0 = 64
  g1[4] = NS;            // tile_dim1 = 12, tile_dim2 = 0
  g1[5] = HW * HW;       // tensor_dim0_stride lo32 = 262144 (channel stride)
  g1[6] = 0;
  g1[7] = 0;
  const i32x4 z4 = {0, 0, 0, 0};
#if defined(__clang_major__) && (__clang_major__ >= 23)
  const i32x8 z8 = {0, 0, 0, 0, 0, 0, 0, 0};
  __builtin_amdgcn_tensor_load_to_lds(g0, g1, z4, z4, z8, 0);
#else
  __builtin_amdgcn_tensor_load_to_lds(g0, g1, z4, z4, 0);
#endif
}

__global__ void __launch_bounds__(128)
CAutomaton_25409026523269_kernel(const float* __restrict__ x,
                                 const float* __restrict__ pw,
                                 const float* __restrict__ pb,
                                 const float* __restrict__ w1,
                                 const float* __restrict__ b1,
                                 const float* __restrict__ w2,
                                 const int*   __restrict__ mask,
                                 float* __restrict__ out) {
  // ---- LDS (62.3 KB) ----
  __shared__ __align__(16) float sX[3 * NS * XCOLS];            // 9504 B halo tile
  __shared__ __align__(16) unsigned short sPW[PC * KP];         // 12 KB
  __shared__ __align__(16) unsigned short sW1[HID * K1];        // 12 KB
  __shared__ __align__(16) unsigned short sW2[16 * K2];         //  3 KB
  __shared__ float sPB[PC];
  __shared__ float sB1[HID];
  __shared__ __align__(16) unsigned short sB[WAVES][16 * KP];   // im2col, reused as h
  __shared__ __align__(16) unsigned short sPerc[WAVES][16 * K1];

  const int tid  = threadIdx.x;
  const int wave = tid >> 5;
  const int lane = tid & 31;

  // ---- convert weights to bf16 LDS (once per block) ----
  for (int i = tid; i < PC * KP; i += blockDim.x) {
    const int m = i >> 7, k = i & (KP - 1);
    sPW[i] = (k < NS * 9) ? bfbits(pw[m * (NS * 9) + k]) : (unsigned short)0;
  }
  for (int i = tid; i < HID * K1; i += blockDim.x) {
    const int m = i >> 6, k = i & (K1 - 1);
    sW1[i] = (k < PC) ? bfbits(w1[m * PC + k]) : (unsigned short)0;
  }
  for (int i = tid; i < 16 * K2; i += blockDim.x) {
    const int m = i / K2, k = i - m * K2;
    sW2[i] = (m < NS) ? bfbits(w2[m * K2 + k]) : (unsigned short)0;
  }
  for (int i = tid; i < PC; i += blockDim.x)  sPB[i] = pb[i];
  for (int i = tid; i < HID; i += blockDim.x) sB1[i] = b1[i];

  unsigned short* B = sB[wave];       // wave-private staging
  unsigned short* P = sPerc[wave];
  const int nn = lane & 15;
  const int mh = (lane >> 4) << 3;

  for (int g = 0; g < GRP; ++g) {
    // 64 contiguous pixels along W per block (W=512 divisible by 64)
    const int base = ((blockIdx.x * GRP + g) * STRIP);
    const int n    = base >> 18;           // / (512*512)
    const int rem  = base & 262143;
    const int r0   = rem >> 9;
    const int cb   = rem & 511;            // strip start column

    // ---- stage halo tile: 3 rows via TDM (wave 0), edge cols via lanes ----
    if (wave == 0) {
      #pragma unroll
      for (int row = 0; row < 3; ++row) {
        const int rr = (r0 - 1 + row) & (HW - 1);
        const float* ga = x + (((n * NS) << 9 | rr) << 9) + cb;
        const unsigned lds_off =
            (unsigned)(unsigned long long)(void*)&sX[row * NS * XCOLS + 1];
        tdm_load_row(ga, lds_off);
      }
      __builtin_amdgcn_s_wait_tensorcnt(0);
    }
    // circular-wrap edge columns: 3 rows x 12 ch x 2 sides = 72 loads
    for (int i = tid; i < 72; i += blockDim.x) {
      const int row  = i / 24;
      const int j    = i - row * 24;
      const int c    = j >> 1;
      const int side = j & 1;
      const int rr   = (r0 - 1 + row) & (HW - 1);
      const int cc   = side ? ((cb + STRIP) & (HW - 1)) : ((cb - 1) & (HW - 1));
      sX[row * NS * XCOLS + c * XCOLS + (side ? (XCOLS - 1) : 0)] =
          x[((((n * NS + c) << 9) + rr) << 9) + cc];
    }
    // prefetch next strip's center row
    if (g + 1 < GRP) {
      const int nb = base + STRIP;
      __builtin_prefetch(x + ((((nb >> 18) * NS) << 9 | ((nb & 262143) >> 9)) << 9)
                           + (nb & 511), 0, 3);
    }
    __syncthreads();

    // ---- build bf16 im2col from LDS: B[p][k], k = c*9 + ky*3 + kx ----
    for (int e = lane; e < 16 * KP; e += 32) {
      const int p = e & 15, k = e >> 4;
      unsigned short val = 0;
      if (k < NS * 9) {
        const int c  = k / 9;
        const int t  = k - c * 9;
        const int ky = t / 3;
        const int kx = t - ky * 3;
        val = bfbits(sX[ky * NS * XCOLS + c * XCOLS + wave * 16 + p + kx]);
      }
      B[p * KP + k] = val;
    }
    __syncthreads();

    // ---- stage P: perc[48x16] = PW[48x128] * B[128x16]  (12 WMMAs) ----
    #pragma unroll
    for (int mt = 0; mt < 3; ++mt) {
      v8f acc = {};
      #pragma unroll
      for (int kc = 0; kc < 4; ++kc) {
        v16bf a = load_frag(sPW, mt * 16 + nn, KP, kc, lane);
        v16bf b = load_frag(B, nn, KP, kc, lane);
        acc = wmma_bf16(a, b, acc);
      }
      const int m0 = mt * 16 + mh;
      #pragma unroll
      for (int r = 0; r < 8; r += 2) {       // packed b32 stores
        union { unsigned int u; unsigned short s[2]; } pk;
        pk.s[0] = bfbits(acc[r]     + sPB[m0 + r]);
        pk.s[1] = bfbits(acc[r + 1] + sPB[m0 + r + 1]);
        *(unsigned int*)&P[nn * K1 + m0 + r] = pk.u;
      }
    }
    // zero K padding 48..63 of perc
    for (int e = lane; e < 16 * 8; e += 32)
      *(unsigned int*)&P[(e >> 3) * K1 + PC + ((e & 7) << 1)] = 0;
    __syncthreads();

    // ---- stage 1: h[96x16] = relu(W1[96x64]*perc + b1)  (12 WMMAs) ----
    #pragma unroll
    for (int mt = 0; mt < 6; ++mt) {
      v8f acc = {};
      #pragma unroll
      for (int kc = 0; kc < 2; ++kc) {
        v16bf a = load_frag(sW1, mt * 16 + nn, K1, kc, lane);
        v16bf b = load_frag(P, nn, K1, kc, lane);
        acc = wmma_bf16(a, b, acc);
      }
      const int m0 = mt * 16 + mh;
      #pragma unroll
      for (int r = 0; r < 8; r += 2) {
        union { unsigned int u; unsigned short s[2]; } pk;
        pk.s[0] = bfbits(fmaxf(acc[r]     + sB1[m0 + r],     0.0f));
        pk.s[1] = bfbits(fmaxf(acc[r + 1] + sB1[m0 + r + 1], 0.0f));
        *(unsigned int*)&B[nn * K2 + m0 + r] = pk.u;   // h reuses B region
      }
    }
    __syncthreads();

    // ---- stage 2: upd[16x16] = W2[16x96] * h[96x16]  (3 WMMAs) ----
    {
      v8f acc = {};
      #pragma unroll
      for (int kc = 0; kc < 3; ++kc) {
        v16bf a = load_frag(sW2, nn, K2, kc, lane);
        v16bf b = load_frag(B, nn, K2, kc, lane);
        acc = wmma_bf16(a, b, acc);
      }
      // masked residual: out = x + upd * mask (D rows 12..15 are padding)
      const int col = cb + wave * 16 + nn;
      #pragma unroll
      for (int r = 0; r < 8; ++r) {
        const int m = mh + r;
        if (m < NS) {
          const int idx = ((((n * NS + m) << 9) + r0) << 9) + col;
          out[idx] = x[idx] + acc[r] * (float)mask[idx];
        }
      }
    }
    __syncthreads();   // protect sX/B/P reuse for the next strip
  }
}

extern "C" void kernel_launch(void* const* d_in, const int* in_sizes, int n_in,
                              void* d_out, int out_size, void* d_ws, size_t ws_size,
                              hipStream_t stream) {
  const float* x    = (const float*)d_in[0];   // [8,12,512,512]
  const float* pw   = (const float*)d_in[1];   // [48,12,3,3] == [48,108]
  const float* pb   = (const float*)d_in[2];   // [48]
  const float* w1   = (const float*)d_in[3];   // [96,48]
  const float* b1   = (const float*)d_in[4];   // [96]
  const float* w2   = (const float*)d_in[5];   // [12,96]
  const int*   mask = (const int*)d_in[6];     // [8,12,512,512]
  float* out        = (float*)d_out;

  // total pixels = 8*512*512 = 2,097,152; per block = 64 * GRP = 256
  const int blocks = (8 * HW * HW) / (STRIP * GRP);   // 8192
  CAutomaton_25409026523269_kernel<<<blocks, 128, 0, stream>>>(
      x, pw, pb, w1, b1, w2, mask, out);
}
